// SSCDNonLModel_37606733643824
// MI455X (gfx1250) — matrix operand order repaired and checked by reading, not compiled, not run on hardware
//
#include <hip/hip_runtime.h>
#include <hip/hip_bf16.h>
#include <stdint.h>

// ---------------------------------------------------------------------------
// CDNA5 / gfx1250 implementation of SSCDNonLModel forward.
// wave32; WMMA 16x16x32 f16->f32 for all 1x1 convolutions (GEMMs);
// async global->LDS copies (ASYNCcnt) for the f16 weight tiles when available.
// ---------------------------------------------------------------------------

typedef __attribute__((ext_vector_type(16))) _Float16 v16h;
typedef __attribute__((ext_vector_type(8)))  float    v8f;

#if __has_builtin(__builtin_amdgcn_global_load_async_to_lds_b32)
#define USE_ASYNC_LDS 1
#endif

typedef __attribute__((address_space(1))) int as1_int;
typedef __attribute__((address_space(3))) int as3_int;

__device__ __forceinline__ void async_wait_all() {
#ifdef USE_ASYNC_LDS
#if __has_builtin(__builtin_amdgcn_s_wait_asynccnt)
    __builtin_amdgcn_s_wait_asynccnt(0);
#else
    asm volatile("s_wait_asynccnt 0" ::: "memory");
#endif
#endif
}

__device__ __forceinline__ float sigmoidf_(float x) {
    return 1.0f / (1.0f + __expf(-x));
}

// ---------------------------------------------------------------------------
// Fold BN (over input channels) + f32->f16 conversion into a zero-padded
// weight matrix Wh[Mpad x Kpad]:   y = W(s*x+t)+b == (W*s)x + (b + W·t)
// bn pointers may be null (s=1, t=0). One block per padded output row.
// ---------------------------------------------------------------------------
__global__ __launch_bounds__(256)
void fold_kernel(const float* __restrict__ W, const float* __restrict__ b,
                 const float* __restrict__ gamma, const float* __restrict__ beta,
                 const float* __restrict__ mean,  const float* __restrict__ var,
                 _Float16* __restrict__ Wh, float* __restrict__ bf,
                 int Cout, int Cin, int Kpad)
{
    __shared__ float red[256];
    const int o = blockIdx.x;
    const int t = threadIdx.x;
    float part = 0.0f;
    for (int c = t; c < Kpad; c += 256) {
        float wv = 0.0f;
        if (o < Cout && c < Cin) {
            float s = 1.0f, sh = 0.0f;
            if (gamma) {
                s  = gamma[c] * rsqrtf(var[c] + 1e-5f);
                sh = beta[c] - mean[c] * s;
            }
            float w = W[o * Cin + c];
            wv = w * s;
            part += w * sh;
        }
        Wh[o * Kpad + c] = (_Float16)wv;
    }
    red[t] = part;
    __syncthreads();
    for (int offs = 128; offs > 0; offs >>= 1) {
        if (t < offs) red[t] += red[t + offs];
        __syncthreads();
    }
    if (t == 0 && o < Cout) bf[o] = b[o] + red[0];
}

// ---------------------------------------------------------------------------
// 1x1-conv GEMM with WMMA:
//   dst[n,co,hw] = act( sum_c Wh[co,c]*src[n,c,hw] + bias[co] )
// Two sources emulate channel-concat (c < C0 -> src0, else src1).
// Block: 256 threads = 8 waves. Tile: 32 Cout x 128 pixels (16/wave).
// Double-buffered LDS, software-pipelined: stage tile k+1 while WMMA on k.
// A tiles (f16 weights, zero-padded) copied with async global->LDS b32 when
// available; B tiles converted f32->f16 with branch-free clamped loads.
// Fragment order in LDS so each lane reads one contiguous v16h:
//   A frag (16-bit A 16x32):  lane = m + 16*((k>>3)&1),  slot = (k&7)|((k&16)>>1)
//   B frag (K x 16 per wave): lane = n + 16*((k>>3)&1),  slot = (k&7)|((k&16)>>1)
// ---------------------------------------------------------------------------
__global__ __launch_bounds__(256)
void gemm1x1_kernel(const float* __restrict__ src0, int C0,
                    const float* __restrict__ src1, int C1,
                    const _Float16* __restrict__ Wh, int Kpad,
                    const float* __restrict__ bias,
                    float* __restrict__ dst, int Cout, int HW, int act)
{
    __shared__ __align__(32) _Float16 ldsA[2][2 * 32 * 16];  // [buf][sub][lane][slot]
    __shared__ __align__(32) _Float16 ldsB[2][8 * 32 * 16];  // [buf][wave][lane][slot]

    const int t    = threadIdx.x;
    const int wv   = t >> 5;
    const int lane = t & 31;
    const int mb   = blockIdx.x * 32;
    const int pb   = blockIdx.y * 128;      // pixel tile base (HW % 128 == 0)
    const int n    = pb / HW;
    const int hw0  = pb - n * HW;
    const int Cin  = C0 + C1;
    const int base0 = n * C0 * HW + hw0;
    const int base1 = n * C1 * HW + hw0;

    // ---- stage A tile (32 rows x 32 K of padded f16 weights), 2 pairs/thread
    auto stageA = [&](int buf, int kb) {
        #pragma unroll
        for (int i = 0; i < 2; ++i) {
            int pi  = t + i * 256;          // pair index 0..511
            int m   = pi >> 4;              // 0..31
            int kk  = (pi & 15) * 2;        // even K
            int sub = m >> 4;
            int alane = (m & 15) | (((kk >> 3) & 1) << 4);
            int aslot = (kk & 7) | ((kk & 16) >> 1);
            const _Float16* g = Wh + (mb + m) * Kpad + kb + kk;       // always in-bounds
            _Float16* l = &ldsA[buf][(sub * 32 + alane) * 16 + aslot];
#ifdef USE_ASYNC_LDS
            __builtin_amdgcn_global_load_async_to_lds_b32(
                (as1_int*)(void*)g, (as3_int*)(void*)l, 0, 0);
#else
            *(uint32_t*)l = *(const uint32_t*)g;
#endif
        }
    };
    // ---- stage B tile (32 K x 128 pixels), branch-free clamped loads ----
    auto stageB = [&](int buf, int kb) {
        #pragma unroll
        for (int i = 0; i < 16; ++i) {
            int idx = t + i * 256;          // 0..4095
            int kk  = idx >> 7;
            int pl  = idx & 127;            // coalesced across threads
            int c   = kb + kk;
            int cA  = (c < C0) ? c : (C0 - 1);
            int cB  = c - C0;
            cB = (cB < 0) ? 0 : cB;
            cB = (cB >= C1) ? (C1 > 0 ? C1 - 1 : 0) : cB;
            const float* pa = src0 + base0 + cA * HW + pl;
            const float* pc = src1 ? (src1 + base1 + cB * HW + pl) : pa;
            const float* p  = (c < C0) ? pa : pc;     // v_cndmask on address
            float v = *p;                              // unconditional, always safe
            v = (c < Cin) ? v : 0.0f;                  // zero the K padding
            int blane = (pl & 15) | (((kk >> 3) & 1) << 4);
            ldsB[buf][(((pl >> 4) * 32 + blane) * 16) + ((kk & 7) | ((kk & 16) >> 1))] = (_Float16)v;
        }
    };

    v8f acc0 = {}, acc1 = {};
    const int nk = (Cin + 31) >> 5;

    stageA(0, 0);
    stageB(0, 0);
    async_wait_all();
    __syncthreads();

    for (int ki = 0; ki < nk; ++ki) {
        const int buf = ki & 1;
        v16h a0 = ((const v16h*)ldsA[buf])[lane];
        v16h a1 = ((const v16h*)ldsA[buf])[32 + lane];
        v16h bf_ = ((const v16h*)ldsB[buf])[wv * 32 + lane];
        if (ki + 1 < nk) {                 // prefetch next tile into other buffer
            stageA(buf ^ 1, (ki + 1) * 32);
            stageB(buf ^ 1, (ki + 1) * 32);
        }
        acc0 = __builtin_amdgcn_wmma_f32_16x16x32_f16(
                   false, a0, false, bf_, (short)0, acc0, false, false);
        acc1 = __builtin_amdgcn_wmma_f32_16x16x32_f16(
                   false, a1, false, bf_, (short)0, acc1, false, false);
        async_wait_all();
        __syncthreads();
    }

    // ---- store: VGPR r -> (M = r + 8*laneHi, N = lane&15) ----
    const int hi   = lane >> 4;
    const int ncol = lane & 15;
    const int p    = hw0 + wv * 16 + ncol;
    #pragma unroll
    for (int r = 0; r < 8; ++r) {
        int row0 = mb + r + 8 * hi;
        int row1 = row0 + 16;
        if (row0 < Cout) {
            float v = acc0[r] + bias[row0];
            if (act) v = sigmoidf_(v);
            dst[(n * Cout + row0) * HW + p] = v;
        }
        if (row1 < Cout) {
            float v = acc1[r] + bias[row1];
            if (act) v = sigmoidf_(v);
            dst[(n * Cout + row1) * HW + p] = v;
        }
    }
}

// ---------------------------------------------------------------------------
// Grouped/depthwise 5x5 conv, pad 2, + bias + sigmoid. Cout == groups here:
// out channel co consumes input channels [co*cig, co*cig+cig).
// ---------------------------------------------------------------------------
__global__ __launch_bounds__(256)
void gconv5_kernel(const float* __restrict__ src, const float* __restrict__ w,
                   const float* __restrict__ b, float* __restrict__ dst,
                   int N, int Cin, int Cout, int cig, int H, int W)
{
    const int HW  = H * W;
    const int idx = blockIdx.x * 256 + threadIdx.x;
    if (idx >= N * Cout * HW) return;
    const int x  = idx % W;
    const int y  = (idx / W) % H;
    const int co = (idx / HW) % Cout;
    const int n  = idx / (HW * Cout);

    float acc = b[co];
    const int ci0 = co * cig;
    for (int ci = 0; ci < cig; ++ci) {
        const float* sp = src + (size_t)(n * Cin + ci0 + ci) * HW;
        const float* wp = w + (co * cig + ci) * 25;
        #pragma unroll
        for (int kh = 0; kh < 5; ++kh) {
            int yy = y + kh - 2;
            if (yy < 0 || yy >= H) continue;
            #pragma unroll
            for (int kw = 0; kw < 5; ++kw) {
                int xx = x + kw - 2;
                if (xx < 0 || xx >= W) continue;
                acc += wp[kh * 5 + kw] * sp[yy * W + xx];
            }
        }
    }
    dst[idx] = sigmoidf_(acc);
}

// ---------------------------------------------------------------------------
// Explicit BN (only b4 needs it: its output xx is reused directly).
// ---------------------------------------------------------------------------
__global__ __launch_bounds__(256)
void bn_kernel(const float* __restrict__ src,
               const float* __restrict__ gamma, const float* __restrict__ beta,
               const float* __restrict__ mean,  const float* __restrict__ var,
               float* __restrict__ dst, int C, int HW, int total)
{
    int idx = blockIdx.x * 256 + threadIdx.x;
    if (idx >= total) return;
    int c = (idx / HW) % C;
    float s = gamma[c] * rsqrtf(var[c] + 1e-5f);
    dst[idx] = (src[idx] - mean[c]) * s + beta[c];
}

// dst = (a + b) * 0.7 + c
__global__ __launch_bounds__(256)
void combine_kernel(const float* __restrict__ a, const float* __restrict__ b,
                    const float* __restrict__ c, float* __restrict__ dst, int total)
{
    int i = blockIdx.x * 256 + threadIdx.x;
    if (i < total) dst[i] = (a[i] + b[i]) * 0.7f + c[i];
}

// ---------------------------------------------------------------------------
// Criss-cross attention, fully fused (energies + joint softmax with column
// diagonal mask + aggregation + 0.6*out + x residual).
// One block per (n, h) row; thread = column w. Row K/V staged in LDS;
// column accesses lane-coalesced and L2-resident. Online softmax, recompute
// second pass (no per-thread 384-float buffer).
// ---------------------------------------------------------------------------
#define CCC 25
__global__ __launch_bounds__(192)
void cca_attn_kernel(const float* __restrict__ q, const float* __restrict__ k,
                     const float* __restrict__ v, const float* __restrict__ xin,
                     float* __restrict__ dst, int H, int W)
{
    __shared__ float Krow[CCC][192];
    __shared__ float Vrow[CCC][192];
    const int HW = H * W;
    const int h  = blockIdx.x % H;
    const int n  = blockIdx.x / H;
    const int w  = threadIdx.x;
    const int base = n * CCC * HW;

    for (int i = threadIdx.x; i < CCC * W; i += blockDim.x) {
        int c = i / W, j = i % W;
        Krow[c][j] = k[base + c * HW + h * W + j];
        Vrow[c][j] = v[base + c * HW + h * W + j];
    }
    __syncthreads();

    float qr[CCC];
    #pragma unroll
    for (int c = 0; c < CCC; ++c) qr[c] = q[base + c * HW + h * W + w];

    // ---- pass 1: running max m and normalizer s over row+col logits ----
    float m = -3.0e38f, s = 0.0f;
    for (int j = 0; j < W; ++j) {
        float e = 0.0f;
        #pragma unroll
        for (int c = 0; c < CCC; ++c) e += qr[c] * Krow[c][j];
        float mn = fmaxf(m, e);
        s = s * __expf(m - mn) + __expf(e - mn);
        m = mn;
    }
    for (int i = 0; i < H; ++i) {
        if (i == h) continue;                    // eye-masked (NEG -> weight 0)
        float e = 0.0f;
        #pragma unroll
        for (int c = 0; c < CCC; ++c) e += qr[c] * k[base + c * HW + i * W + w];
        float mn = fmaxf(m, e);
        s = s * __expf(m - mn) + __expf(e - mn);
        m = mn;
    }

    // ---- pass 2: weighted aggregation (recompute logits) ----
    float out[CCC];
    #pragma unroll
    for (int c = 0; c < CCC; ++c) out[c] = 0.0f;
    for (int j = 0; j < W; ++j) {
        float e = 0.0f;
        #pragma unroll
        for (int c = 0; c < CCC; ++c) e += qr[c] * Krow[c][j];
        float a = __expf(e - m);
        #pragma unroll
        for (int c = 0; c < CCC; ++c) out[c] += a * Vrow[c][j];
    }
    for (int i = 0; i < H; ++i) {
        if (i == h) continue;
        float e = 0.0f;
        #pragma unroll
        for (int c = 0; c < CCC; ++c) e += qr[c] * k[base + c * HW + i * W + w];
        float a = __expf(e - m);
        #pragma unroll
        for (int c = 0; c < CCC; ++c) out[c] += a * v[base + c * HW + i * W + w];
    }

    const float inv = 1.0f / s;
    #pragma unroll
    for (int c = 0; c < CCC; ++c) {
        int o = base + c * HW + h * W + w;
        dst[o] = 0.6f * out[c] * inv + xin[o];
    }
}

// ---------------------------------------------------------------------------
// Host orchestration
// ---------------------------------------------------------------------------
extern "C" void kernel_launch(void* const* d_in, const int* in_sizes, int n_in,
                              void* d_out, int out_size, void* d_ws, size_t ws_size,
                              hipStream_t stream)
{
    (void)in_sizes; (void)n_in; (void)out_size; (void)ws_size;

    const int Hh = 192, Ww = 192, HW = Hh * Ww, Nn = 2;
    const int NHW = Nn * HW;

    auto f  = [&](int i) { return (const float*)d_in[i]; };
    auto kp = [](int c) { return (c + 31) & ~31; };   // K padded to 32
    auto mp = [](int c) { return (c + 31) & ~31; };   // M padded to 32
    const float* x = f(0);

    // ---- workspace carve-out ----
    char* ws = (char*)d_ws;
    size_t off = 0;
    auto alloc = [&](size_t bytes) -> void* {
        void* p = ws + off;
        off += (bytes + 255) & ~(size_t)255;
        return p;
    };
    const size_t big = (size_t)Nn * 100 * HW;
    const size_t sml = (size_t)Nn * 25 * HW;

    float* tA   = (float*)alloc(big * 4);
    float* tB   = (float*)alloc(big * 4);
    float* tC   = (float*)alloc(big * 4);
    float* out2 = (float*)alloc(sml * 4);
    float* xx   = (float*)alloc(sml * 4);
    float* qb   = (float*)alloc(sml * 4);
    float* kb   = (float*)alloc(sml * 4);
    float* vb   = (float*)alloc(sml * 4);
    float* y1   = (float*)alloc(sml * 4);
    float* nl2  = (float*)alloc(sml * 4);
    float* nl3  = (float*)alloc(sml * 4);

    _Float16* WhA = (_Float16*)alloc((size_t)mp(100) * kp(200) * 2);
    _Float16* WhB = (_Float16*)alloc((size_t)mp(100) * kp(300) * 2);
    _Float16* WhD = (_Float16*)alloc((size_t)mp(100) * kp(50)  * 2);
    _Float16* WhE = (_Float16*)alloc((size_t)mp(100) * kp(125) * 2);
    _Float16* WhF = (_Float16*)alloc((size_t)mp(17)  * kp(125) * 2);
    _Float16* Whcc[6];
    for (int i = 0; i < 6; ++i) Whcc[i] = (_Float16*)alloc((size_t)mp(25) * kp(25) * 2);

    float* bfA = (float*)alloc(100 * 4);
    float* bfB = (float*)alloc(100 * 4);
    float* bfD = (float*)alloc(100 * 4);
    float* bfE = (float*)alloc(100 * 4);
    float* bfF = (float*)alloc(17 * 4);
    float* bfcc[6];
    for (int i = 0; i < 6; ++i) bfcc[i] = (float*)alloc(25 * 4);

    // ---- fold BN into 1x1-conv weights (zero-padded, f16) ----
    fold_kernel<<<mp(100), 256, 0, stream>>>(f(5),  f(6),  f(1),  f(2),  f(3),  f(4),  WhA, bfA, 100, 200, kp(200));
    fold_kernel<<<mp(100), 256, 0, stream>>>(f(13), f(14), f(9),  f(10), f(11), f(12), WhB, bfB, 100, 300, kp(300));
    fold_kernel<<<mp(100), 256, 0, stream>>>(f(37), f(38), f(33), f(34), f(35), f(36), WhD, bfD, 100, 50,  kp(50));
    fold_kernel<<<mp(100), 256, 0, stream>>>(f(45), f(46), f(41), f(42), f(43), f(44), WhE, bfE, 100, 125, kp(125));
    fold_kernel<<<mp(17),  256, 0, stream>>>(f(49), f(50), nullptr, nullptr, nullptr, nullptr, WhF, bfF, 17, 125, kp(125));
    for (int i = 0; i < 6; ++i) {   // cc2: 21..26, cc3: 27..32
        fold_kernel<<<mp(25), 256, 0, stream>>>(f(21 + 2 * i), f(22 + 2 * i),
                                                nullptr, nullptr, nullptr, nullptr,
                                                Whcc[i], bfcc[i], 25, 25, kp(25));
    }

    auto gemm = [&](const float* s0, int C0, const float* s1, int C1,
                    const _Float16* Wh, int Kpad, const float* bias,
                    float* dst, int Cout, int act) {
        dim3 g((Cout + 31) / 32, NHW / 128);
        gemm1x1_kernel<<<g, 256, 0, stream>>>(s0, C0, s1, C1, Wh, Kpad, bias, dst, Cout, HW, act);
    };
    auto gconv = [&](const float* src, const float* w, const float* b, float* dst,
                     int Cin, int Cout, int cig) {
        int total = Nn * Cout * HW;
        gconv5_kernel<<<(total + 255) / 256, 256, 0, stream>>>(src, w, b, dst, Nn, Cin, Cout, cig, Hh, Ww);
    };
    auto cca = [&](const float* src, int pi, float* dst) {  // pi: 0 = cc2, 3 = cc3
        gemm(src, 25, nullptr, 0, Whcc[pi + 0], kp(25), bfcc[pi + 0], qb, 25, 0);
        gemm(src, 25, nullptr, 0, Whcc[pi + 1], kp(25), bfcc[pi + 1], kb, 25, 0);
        gemm(src, 25, nullptr, 0, Whcc[pi + 2], kp(25), bfcc[pi + 2], vb, 25, 0);
        cca_attn_kernel<<<Nn * Hh, 192, 0, stream>>>(qb, kb, vb, src, dst, Hh, Ww);
    };

    // ---- main pipeline ----
    gemm(x, 200, nullptr, 0, WhA, kp(200), bfA, tA, 100, 1);   // sig(con1(bn1(x)))
    gconv(tA, f(7), f(8), tB, 100, 100, 1);                    // out1 = sig(cond1)
    gemm(tB, 100, x, 200, WhB, kp(300), bfB, tC, 100, 1);      // sig(con2(bn2([out1,x])))
    gconv(tC, f(15), f(16), out2, 100, 25, 4);                 // out2 = sig(cond2)

    {   // xx = bn(out2, b4)
        int total = Nn * 25 * HW;
        bn_kernel<<<(total + 255) / 256, 256, 0, stream>>>(out2, f(17), f(18), f(19), f(20),
                                                           xx, 25, HW, total);
    }

    cca(xx, 0, y1);  cca(y1, 0, nl2);                          // nl2 = cca(cca(xx, cc2), cc2)
    cca(xx, 3, y1);  cca(y1, 3, nl3);                          // nl3 = cca(cca(xx, cc3), cc3)
    {   // y1 = (nl2 + nl3)*0.7 + xx
        int total = Nn * 25 * HW;
        combine_kernel<<<(total + 255) / 256, 256, 0, stream>>>(nl2, nl3, xx, y1, total);
    }

    gemm(xx, 25, y1, 25, WhD, kp(50), bfD, tA, 100, 1);        // sig(con4(bcat([xx, nl2])))
    gconv(tA, f(39), f(40), tB, 100, 100, 1);                  // out4 = sig(cond4)
    gemm(tB, 100, out2, 25, WhE, kp(125), bfE, tC, 100, 1);    // sig(con5(b5([out4, out2])))
    gconv(tC, f(47), f(48), tA, 100, 100, 1);                  // out5 = sig(cond5)
    gemm(tA, 100, out2, 25, WhF, kp(125), bfF, (float*)d_out, 17, 0); // con6([out5,out2])
}
